// EEGCSNN_89618787598579
// MI455X (gfx1250) — matrix operand order repaired and checked by reading, not compiled, not run on hardware
//
#include <hip/hip_runtime.h>

typedef __attribute__((ext_vector_type(16))) _Float16 v16h;
typedef __attribute__((ext_vector_type(4)))  _Float16 v4h;
typedef __attribute__((ext_vector_type(8)))  float    v8f;
typedef __attribute__((ext_vector_type(4)))  float    v4f;
typedef __attribute__((ext_vector_type(4)))  unsigned int v4u;
typedef __attribute__((ext_vector_type(4)))  int      v4ii;
typedef __attribute__((ext_vector_type(8)))  int      v8ii;

#define BETA 0.9f
#define THR  1.0f
#define T_STEPS 25

#if __has_builtin(__builtin_amdgcn_tensor_load_to_lds) && \
    __has_builtin(__builtin_amdgcn_s_wait_tensorcnt)
#define USE_TDM 1
#endif

// ---- workspace layout (bytes) ----
// cur1 : [128,8,64,64] f32   16,777,216 B
// m1   : [128,8,64,64] f32   16,777,216 B
// s1h  : [128,8,64,64] f16    8,388,608 B
// m2   : [128,32,32,16] f32 (channels-last)  8,388,608 B
// s2h  : [128,32,32,16] f16 (channels-last)  4,194,304 B
// m3   : [128,4] f32              2,048 B
// w2h  : [16,96] f16 (K padded)   3,072 B
// wlh  : [4,16384] f16          131,072 B
#define OFF_CUR1 ((size_t)0)
#define OFF_M1   ((size_t)16777216)
#define OFF_S1H  ((size_t)33554432)
#define OFF_M2   ((size_t)41943040)
#define OFF_S2H  ((size_t)50331648)
#define OFF_M3   ((size_t)54525952)
#define OFF_W2H  ((size_t)54528000)
#define OFF_WLH  ((size_t)54531072)

// ---------------------------------------------------------------------------
// Init: zero membrane state every call (determinism), pack W2 into padded
// K=96 f16 layout k = (ty*3+tx)*8 + c, convert Wl to f16.
// ---------------------------------------------------------------------------
__global__ void k_init(const float* __restrict__ W2, const float* __restrict__ Wl,
                       float* __restrict__ m1, float* __restrict__ m2,
                       float* __restrict__ m3,
                       _Float16* __restrict__ w2h, _Float16* __restrict__ wlh) {
    int i = blockIdx.x * blockDim.x + threadIdx.x;
    if (i < 4194304) m1[i] = 0.f;
    if (i < 2097152) m2[i] = 0.f;
    if (i < 512)     m3[i] = 0.f;
    if (i < 65536)   wlh[i] = (_Float16)Wl[i];
    if (i < 1536) {
        int n = i / 96, k = i % 96;
        float v = 0.f;
        if (k < 72) {
            int tap = k >> 3, c = k & 7;
            int ty = tap / 3, tx = tap - ty * 3;
            v = W2[((n * 8 + c) * 3 + ty) * 3 + tx];
        }
        w2h[i] = (_Float16)v;
    }
}

// ---------------------------------------------------------------------------
// conv1 (1->8ch, 3x3 pad1) + 2x2 maxpool, f32 direct (runs once; tiny).
// ---------------------------------------------------------------------------
__global__ void k_conv1(const float* __restrict__ x, const float* __restrict__ W1,
                        const float* __restrict__ b1, float* __restrict__ cur1) {
    int i = blockIdx.x * blockDim.x + threadIdx.x;
    if (i >= 128 * 8 * 64 * 64) return;
    int px = i & 63, py = (i >> 6) & 63, co = (i >> 12) & 7, b = i >> 15;
    float w[9];
#pragma unroll
    for (int j = 0; j < 9; ++j) w[j] = W1[co * 9 + j];
    const float* xb = x + (size_t)b * 16384;
    float best = -1e30f;
#pragma unroll
    for (int dy = 0; dy < 2; ++dy)
#pragma unroll
        for (int dx = 0; dx < 2; ++dx) {
            int y = 2 * py + dy, xx = 2 * px + dx;
            float acc = 0.f;
#pragma unroll
            for (int ty = 0; ty < 3; ++ty)
#pragma unroll
                for (int tx = 0; tx < 3; ++tx) {
                    int iy = y + ty - 1, ix = xx + tx - 1;
                    if (iy >= 0 && iy < 128 && ix >= 0 && ix < 128)
                        acc += xb[iy * 128 + ix] * w[ty * 3 + tx];
                }
            best = fmaxf(best, acc);
        }
    cur1[i] = best + b1[co];
}

// ---------------------------------------------------------------------------
// LIF layer 1, vectorized x4 (b128 loads, b64 f16 spike store).
// m1 = 0.9*m1 + cur1 - (m1_old>1); spike stored as exact f16 0/1.
// ---------------------------------------------------------------------------
__global__ void k_lif1(const v4f* __restrict__ cur1, v4f* __restrict__ m1,
                       v4h* __restrict__ s1h) {
    int i = blockIdx.x * blockDim.x + threadIdx.x;
    if (i >= 1048576) return;
    v4f mo = m1[i], cu = cur1[i];
    v4f mn;
    v4h sp;
#pragma unroll
    for (int j = 0; j < 4; ++j) {
        float s = mo[j] > THR ? 1.f : 0.f;
        mn[j] = BETA * mo[j] + cu[j] - s;
        sp[j] = (mn[j] > THR) ? (_Float16)1.f : (_Float16)0.f;
    }
    m1[i] = mn;
    s1h[i] = sp;
}

// ---------------------------------------------------------------------------
// conv2 (8->16ch, 3x3 pad1) as implicit GEMM via v_wmma_f32_16x16x32_f16,
// fused 2x2 maxpool + bias + LIF2.  One block = one half (32 pre-pool rows)
// of one batch image.  Spike tile staged in LDS (stride-80 rows, zero halo)
// by the Tensor Data Mover (one 2D descriptor per input channel, hardware
// row padding 32dw + 8dw), falling back to manual copy if TDM builtin absent.
// GEMM: M=16 pixels, N=16 out channels, K=72 (tap-major, 8 ch per tap) padded to 96.
// ---------------------------------------------------------------------------
__global__ __launch_bounds__(256) void k_conv2(
        const _Float16* __restrict__ s1h, const _Float16* __restrict__ w2h,
        const float* __restrict__ b2, float* __restrict__ m2,
        _Float16* __restrict__ s2h) {
    // [c][row h0-1 .. h0+32][80], interior columns 8..71 ; 43,520 B
    alignas(16) __shared__ _Float16 smem[8 * 34 * 80];
    int b = blockIdx.x >> 1;
    int half = blockIdx.x & 1;
    int h0 = half * 32;
    int tid = threadIdx.x;
    int lane = tid & 31;
    int wave = tid >> 5;

    for (int j = tid; j < 8 * 34 * 80; j += 256) smem[j] = (_Float16)0.f;
    __syncthreads();

#ifdef USE_TDM
    {
        // Wave w DMAs channel c = w: 2D tile 64 x 33 rows (clipped to image),
        // hardware inserts 8 dwords of LDS pad after each 32-dword (64 f16) row.
        int c = wave;
        int ystart = (h0 == 0) ? 0 : (h0 - 1);
        int rloc0  = ystart - (h0 - 1);          // 1 if h0==0 else 0
        unsigned long long gaddr = (unsigned long long)(size_t)s1h +
            ((((unsigned long long)b * 8 + (unsigned)c) * 64 + (unsigned)ystart) * 64ull) * 2ull;
        unsigned lds_base =
            (unsigned)(size_t)((__attribute__((address_space(3))) _Float16*)smem);
        unsigned lds_addr = lds_base + (unsigned)((c * 34 + rloc0) * 160 + 16);

        v4u g0;
        g0[0] = 1u;                                   // count=1 (valid user D#)
        g0[1] = lds_addr;                             // lds_addr      [63:32]
        g0[2] = (unsigned)(gaddr & 0xffffffffu);      // global_addr   [95:64]
        g0[3] = (unsigned)((gaddr >> 32) & 0x1ffffffu) | (2u << 30); // + type=2

        v8ii g1;
        g1[0] = (1 << 16)     // data_size = 2 bytes
              | (1 << 20)     // pad_enable
              | (4 << 22)     // pad_interval: 32 dwords (128 B row)
              | (7 << 25);    // pad_amount:   8 dwords  (16 f16 -> stride 80)
        g1[1] = (int)(64u << 16);     // tensor_dim0 = 64        [79:48] lo
        g1[2] = (int)(33u << 16);     // dim0 hi=0 | tensor_dim1 = 33
        g1[3] = (int)(64u << 16);     // dim1 hi=0 | tile_dim0 = 64
        g1[4] = 33;                   // tile_dim1 = 33 rows, tile_dim2 = 0
        g1[5] = 64;                   // tensor_dim0_stride = 64 elements
        g1[6] = 0;
        g1[7] = 0;
        v4ii g2 = {0, 0, 0, 0};
        v4ii g3 = {0, 0, 0, 0};
        v8ii g4 = {0, 0, 0, 0, 0, 0, 0, 0};
        // 6-arg toolchain variant (clang-23 / therock-10.0 headers).
        __builtin_amdgcn_tensor_load_to_lds(g0, g1, g2, g3, g4, 0);
        __builtin_amdgcn_s_wait_tensorcnt(0);
    }
#else
    for (int j = tid; j < 8 * 34 * 64; j += 256) {
        int c = j / (34 * 64);
        int rem = j % (34 * 64);
        int r = rem >> 6;
        int xx = rem & 63;
        int iy = h0 - 1 + r;
        if (iy >= 0 && iy < 64)
            smem[(c * 34 + r) * 80 + 8 + xx] =
                s1h[(((size_t)b * 8 + c) * 64 + iy) * 64 + xx];
    }
#endif
    __syncthreads();

    int n = lane & 15;   // matrix column (out channel) for B/D, row (pixel) for A
    int g = lane >> 4;

    // B fragments (32x16 f16): lane n = col, K = (v<4?0:16) + g*8 + 2*(v&3) + h
    v16h bf[3];
#pragma unroll
    for (int w = 0; w < 3; ++w)
#pragma unroll
        for (int v = 0; v < 8; ++v)
#pragma unroll
            for (int h = 0; h < 2; ++h)
                bf[w][2 * v + h] = w2h[n * 96 + w * 32 + ((v < 4) ? 0 : 16) + g * 8 + 2 * (v & 3) + h];

    float biasn = b2[n];

    // 16 pooled rows * 4 x-tiles = 64 tiles per block over 8 waves
    for (int iter = wave; iter < 64; iter += 8) {
        int qloc = iter >> 2;       // local pooled row
        int x0 = (iter & 3) << 4;   // pre-pool x tile base
        int x = x0 + n;             // this lane's A-row pixel x
        v8f d[2];
#pragma unroll
        for (int rr = 0; rr < 2; ++rr) {
            int yl = 2 * qloc + rr;  // pre-pool y - h0
            v8f acc = {};
#pragma unroll
            for (int w = 0; w < 3; ++w) {
                v16h a;
#pragma unroll
                for (int v = 0; v < 8; ++v)
#pragma unroll
                    for (int h = 0; h < 2; ++h) {
                        // A (16x32 f16) layout: K = (v<4?0:16) + g*8 + 2*(v&3) + h
                        int kk = w * 32 + ((v < 4) ? 0 : 16) + g * 8 + 2 * (v & 3) + h;
                        _Float16 val = (_Float16)0.f;
                        if (kk < 72) {
                            int tap = kk >> 3, c = kk & 7;
                            int ty = tap / 3, tx = tap - ty * 3;
                            val = smem[(c * 34 + yl + ty) * 80 + (x + tx + 7)];
                        }
                        a[2 * v + h] = val;
                    }
                acc = __builtin_amdgcn_wmma_f32_16x16x32_f16(
                    false, a, false, bf[w], (short)0, acc, false, false);
            }
            d[rr] = acc;
        }
        // 2x2 maxpool (+bias) fully in-register:
        // D layout: lane n = channel, VGPR r -> pixel m = r + g*8.
        int q = (h0 >> 1) + qloc;   // global pooled row
#pragma unroll
        for (int i = 0; i < 4; ++i) {
            float p = fmaxf(fmaxf(d[0][2 * i], d[0][2 * i + 1]),
                            fmaxf(d[1][2 * i], d[1][2 * i + 1])) + biasn;
            int px = (x0 >> 1) + i + g * 4;
            int idx = (((b * 32) + q) * 32 + px) * 16 + n;   // channels-last, coalesced
            float mo = m2[idx];
            float sp = mo > THR ? 1.f : 0.f;
            float mn = BETA * mo + p - sp;
            m2[idx] = mn;
            s2h[idx] = (mn > THR) ? (_Float16)1.f : (_Float16)0.f;
        }
    }
}

// ---------------------------------------------------------------------------
// FC (128x16384 @ 16384x4, N padded to 16) via WMMA, K split over 8 waves,
// LDS reduction, fused LIF3 -> d_out[t].
// s2h is channels-last; NCHW flatten index k maps to (k&1023)*16 + (k>>10).
// ---------------------------------------------------------------------------
__global__ __launch_bounds__(256) void k_fc(
        const _Float16* __restrict__ s2h, const _Float16* __restrict__ wlh,
        const float* __restrict__ bl, const float* __restrict__ beta3p,
        float* __restrict__ m3, float* __restrict__ out, int step) {
    __shared__ float red[8 * 256];
    int tid = threadIdx.x;
    int lane = tid & 31, wave = tid >> 5;
    int b_base = blockIdx.x * 16;
    int mrow = lane & 15;
    int g = lane >> 4;
    const _Float16* arow = s2h + (size_t)(b_base + mrow) * 16384;

    v8f acc = {};
    int kw = wave * 2048;
    for (int j = 0; j < 64; ++j) {
        int kb = kw + j * 32;
        v16h a, bb;
#pragma unroll
        for (int v = 0; v < 8; ++v)
#pragma unroll
            for (int h = 0; h < 2; ++h) {
                int ka = kb + ((v < 4) ? 0 : 16) + g * 8 + 2 * (v & 3) + h;
                a[2 * v + h] = arow[(ka & 1023) * 16 + (ka >> 10)];
                int kv = kb + ((v < 4) ? 0 : 16) + g * 8 + 2 * (v & 3) + h;
                bb[2 * v + h] = (mrow < 4) ? wlh[mrow * 16384 + kv] : (_Float16)0.f;
            }
        acc = __builtin_amdgcn_wmma_f32_16x16x32_f16(
            false, a, false, bb, (short)0, acc, false, false);
    }
#pragma unroll
    for (int r = 0; r < 8; ++r) red[wave * 256 + lane * 8 + r] = acc[r];
    __syncthreads();

    // 256 threads each own one (lane, r) D slot; sum the 8 wave partials.
    float s = 0.f;
#pragma unroll
    for (int w = 0; w < 8; ++w) s += red[w * 256 + tid];
    int L = tid >> 3, r = tid & 7;
    int nn = L & 15;
    int m = r + (L >> 4) * 8;
    if (nn < 4) {
        float b3 = fminf(fmaxf(beta3p[0], 0.f), 1.f);   // snntorch beta clamp
        int bidx = b_base + m;
        float cur = s + bl[nn];
        int mi = bidx * 4 + nn;
        float mo = m3[mi];
        float sp = mo > THR ? 1.f : 0.f;
        float mn = b3 * mo + cur - sp;
        m3[mi] = mn;
        out[(step * 128 + bidx) * 4 + nn] = (mn > THR) ? 1.f : 0.f;
    }
}

// ---------------------------------------------------------------------------
extern "C" void kernel_launch(void* const* d_in, const int* in_sizes, int n_in,
                              void* d_out, int out_size, void* d_ws, size_t ws_size,
                              hipStream_t stream) {
    const float* x     = (const float*)d_in[0];
    const float* W1    = (const float*)d_in[1];
    const float* b1    = (const float*)d_in[2];
    const float* W2    = (const float*)d_in[3];
    const float* b2    = (const float*)d_in[4];
    const float* Wl    = (const float*)d_in[5];
    const float* bl    = (const float*)d_in[6];
    const float* beta3 = (const float*)d_in[7];

    char* ws = (char*)d_ws;
    float*    cur1 = (float*)   (ws + OFF_CUR1);
    float*    m1   = (float*)   (ws + OFF_M1);
    _Float16* s1h  = (_Float16*)(ws + OFF_S1H);
    float*    m2   = (float*)   (ws + OFF_M2);
    _Float16* s2h  = (_Float16*)(ws + OFF_S2H);
    float*    m3   = (float*)   (ws + OFF_M3);
    _Float16* w2h  = (_Float16*)(ws + OFF_W2H);
    _Float16* wlh  = (_Float16*)(ws + OFF_WLH);

    k_init <<<16384, 256, 0, stream>>>(W2, Wl, m1, m2, m3, w2h, wlh);
    k_conv1<<<16384, 256, 0, stream>>>(x, W1, b1, cur1);

    for (int t = 0; t < T_STEPS; ++t) {
        k_lif1 <<<4096, 256, 0, stream>>>((const v4f*)cur1, (v4f*)m1, (v4h*)s1h);
        k_conv2<<<256,  256, 0, stream>>>(s1h, w2h, b2, m2, s2h);
        k_fc   <<<8,    256, 0, stream>>>(s2h, wlh, bl, beta3, m3,
                                          (float*)d_out, t);
    }
}
